// LearnableDelayRNN_20641612825357
// MI455X (gfx1250) — matrix operand order repaired and checked by reading, not compiled, not run on hardware
//
#include <hip/hip_runtime.h>
#include <hip/hip_bf16.h>

typedef __attribute__((ext_vector_type(16))) __bf16 v16bf;
typedef __attribute__((ext_vector_type(8)))  __bf16 v8bf;
typedef __attribute__((ext_vector_type(8)))  float  v8f;
typedef __attribute__((ext_vector_type(4)))  float  v4f;

#define WMMA_BF16(A, Bm, C) \
  __builtin_amdgcn_wmma_f32_16x16x32_bf16(false, (A), false, (Bm), (short)0, (C), false, false)

// Problem constants (B=64, T=256, I=256, H=512, O=256, D=16)
#define BB 64
#define TT 256
#define II 256
#define HH 512
#define OO 256
#define NSLOT 17

static __device__ __forceinline__ void split_bf16(float v, __bf16& hi, __bf16& lo) {
  hi = (__bf16)v;
  lo = (__bf16)(v - (float)hi);
}

// Issue a 32B/lane async copy (global -> LDS), two b128 transfers per lane.
// Same immediate offset applies to both global and LDS addresses (ISA 10.7/15.18).
static __device__ __forceinline__ void stage_frag_async(const __bf16* gsrc, unsigned lds_off,
                                                        int lane) {
  unsigned long long ga = (unsigned long long)(uintptr_t)gsrc + (unsigned)lane * 32u;
  unsigned la = lds_off + (unsigned)lane * 32u;
  asm volatile("global_load_async_to_lds_b128 %0, %1, off" :: "v"(la), "v"(ga) : "memory");
  asm volatile("global_load_async_to_lds_b128 %0, %1, off offset:16" :: "v"(la), "v"(ga) : "memory");
}

// ---------------- precompute kernels ----------------

__global__ void k_zero(float* __restrict__ p, int n) {
  int i = blockIdx.x * 256 + threadIdx.x;
  if (i < n) p[i] = 0.0f;
}

// Wa [H=512][I=256] -> B-fragment packed (nt:32, kc:8, lane:32, e:16), hi/lo
__global__ void k_pack_wa(const float* __restrict__ wa, __bf16* __restrict__ hi, __bf16* __restrict__ lo) {
  int tid = blockIdx.x * 256 + threadIdx.x;
  if (tid >= 32 * 8 * 32 * 16) return;
  int e = tid & 15, lane = (tid >> 4) & 31, kc = (tid >> 9) & 7, nt = tid >> 12;
  int n = nt * 16 + (lane & 15);                 // H index (N dim)
  int k = kc * 32 + (e & 7) + ((e >> 3) << 4) + ((lane >> 4) << 3);  // I index (K dim)
  float v = wa[n * II + k];
  __bf16 h, l; split_bf16(v, h, l);
  hi[tid] = h; lo[tid] = l;
}

// We [O=256][H=512] -> packed (nt:16, kc:16, lane:32, e:16), hi/lo
__global__ void k_pack_we(const float* __restrict__ we, __bf16* __restrict__ hi, __bf16* __restrict__ lo) {
  int tid = blockIdx.x * 256 + threadIdx.x;
  if (tid >= 16 * 16 * 32 * 16) return;
  int e = tid & 15, lane = (tid >> 4) & 31, kc = (tid >> 9) & 15, nt = tid >> 13;
  int n = nt * 16 + (lane & 15);                 // O index
  int k = kc * 32 + (e & 7) + ((e >> 3) << 4) + ((lane >> 4) << 3);  // H index
  float v = we[n * HH + k];
  __bf16 h, l; split_bf16(v, h, l);
  hi[tid] = h; lo[tid] = l;
}

// cm[d][h_out][h_in] = relu(0.125 - 0.015625*|d - clip(tau,1,16)|) * lateral
// packed as B = cm[d]^T : (dm1:16, nt:32, kc:16, lane:32, e:16), hi/lo
__global__ void k_pack_cm(const float* __restrict__ lateral, const float* __restrict__ tau,
                          __bf16* __restrict__ hi, __bf16* __restrict__ lo) {
  int tid = blockIdx.x * 256 + threadIdx.x;
  if (tid >= 16 * 32 * 16 * 32 * 16) return;
  int e = tid & 15, lane = (tid >> 4) & 31, kc = (tid >> 9) & 15;
  int nt = (tid >> 13) & 31, dm1 = tid >> 18;
  int n = nt * 16 + (lane & 15);                 // h_out (N dim)
  int k = kc * 32 + (e & 7) + ((e >> 3) << 4) + ((lane >> 4) << 3);  // h_in  (K dim)
  float tc = tau[n * HH + k];
  tc = fminf(fmaxf(tc, 1.0f), 16.0f);
  float d = (float)(dm1 + 1);
  float c = 0.125f - 0.015625f * fabsf(d - tc);
  c = fmaxf(c, 0.0f);
  float v = c * lateral[n * HH + k];
  __bf16 h, l; split_bf16(v, h, l);
  hi[tid] = h; lo[tid] = l;
}

// ---------------- per-step kernels ----------------

// h = tanh(x_t @ Wa^T + b_a + hdel[slot]); zero slot after read; emit h as bf16 hi/lo.
__global__ __launch_bounds__(32) void k_afferent(
    const float* __restrict__ x, const __bf16* __restrict__ wa_hi, const __bf16* __restrict__ wa_lo,
    const float* __restrict__ ba, float* __restrict__ hdel,
    __bf16* __restrict__ h_hi, __bf16* __restrict__ h_lo, int t, int slot) {
  const int nt = blockIdx.x;        // 0..31 (N tiles over H)
  const int mt = blockIdx.y;        // 0..3  (M tiles over B)
  const int lane = threadIdx.x;     // 0..31
  const int half = lane >> 4;       // lane group
  const int row = mt * 16 + (lane & 15);   // batch index
  const float* xrow = x + ((size_t)row * TT + t) * II;

  v8f acc = {};
  #pragma unroll
  for (int kc = 0; kc < 8; ++kc) {
    const int k0 = kc * 32 + (half ? 8 : 0);
    v4f f0 = *(const v4f*)(xrow + k0);
    v4f f1 = *(const v4f*)(xrow + k0 + 4);
    v4f f2 = *(const v4f*)(xrow + k0 + 16);
    v4f f3 = *(const v4f*)(xrow + k0 + 20);
    v16bf ah, al;
    #pragma unroll
    for (int i = 0; i < 4; ++i) { __bf16 h, l; split_bf16(f0[i], h, l); ah[i]      = h; al[i]      = l; }
    #pragma unroll
    for (int i = 0; i < 4; ++i) { __bf16 h, l; split_bf16(f1[i], h, l); ah[4 + i]  = h; al[4 + i]  = l; }
    #pragma unroll
    for (int i = 0; i < 4; ++i) { __bf16 h, l; split_bf16(f2[i], h, l); ah[8 + i]  = h; al[8 + i]  = l; }
    #pragma unroll
    for (int i = 0; i < 4; ++i) { __bf16 h, l; split_bf16(f3[i], h, l); ah[12 + i] = h; al[12 + i] = l; }
    const size_t boff = ((size_t)(nt * 8 + kc) * 32 + lane) * 16;
    v16bf bh = *(const v16bf*)(wa_hi + boff);
    v16bf bl = *(const v16bf*)(wa_lo + boff);
    acc = WMMA_BF16(ah, bh, acc);
    acc = WMMA_BF16(al, bh, acc);
    acc = WMMA_BF16(ah, bl, acc);
  }
  float* slotp = hdel + (size_t)slot * BB * HH;
  const int n = nt * 16 + (lane & 15);
  const float bias = ba[n];
  #pragma unroll
  for (int r = 0; r < 8; ++r) {
    const int m = mt * 16 + r + 8 * half;
    const size_t idx = (size_t)m * HH + n;
    float pre = acc[r] + bias + slotp[idx];
    slotp[idx] = 0.0f;                       // consume slot (reference masks it)
    float hv = tanhf(pre);
    __bf16 hh, hl; split_bf16(hv, hh, hl);
    h_hi[idx] = hh; h_lo[idx] = hl;
  }
}

// hdel[(t+d)%17] += h @ cm[d]^T, for d = blockIdx.y+1.
// 4 waves (one per M tile) share B fragments, staged into LDS with
// double-buffered async global->LDS copies issued by waves 0 (hi) and 1 (lo).
__global__ __launch_bounds__(128) void k_lateral(
    const __bf16* __restrict__ h_hi, const __bf16* __restrict__ h_lo,
    const __bf16* __restrict__ cm_hi, const __bf16* __restrict__ cm_lo,
    float* __restrict__ hdel, int t) {
  __shared__ __align__(32) __bf16 sB[2][2][512];   // [buf][hi/lo][32 lanes x 16 e]

  const int nt = blockIdx.x;            // 0..31
  const int dm1 = blockIdx.y;           // 0..nd-1
  const int wid = threadIdx.x >> 5;     // 0..3 -> M tile
  const int lane = threadIdx.x & 31;
  const int half = lane >> 4;
  const int row = wid * 16 + (lane & 15);
  const __bf16* arh = h_hi + (size_t)row * HH;
  const __bf16* arl = h_lo + (size_t)row * HH;

  const unsigned lds0 = (unsigned)(uintptr_t)&sB[0][0][0];
  const size_t fbase = (((size_t)dm1 * 32 + nt) * 16) * 512;  // element offset of kc=0 fragment

  // prologue: stage kc=0 into buffer 0 (wave0 -> hi, wave1 -> lo)
  if (wid < 2) {
    const __bf16* g = (wid == 0 ? cm_hi : cm_lo) + fbase;
    stage_frag_async(g, lds0 + (unsigned)wid * 1024u, lane);
  }

  v8f acc = {};
  for (int kc = 0; kc < 16; ++kc) {
    const int buf = kc & 1;
    if (wid < 2) {
      if (kc < 15) {  // stage next fragment into the other buffer
        const __bf16* g = (wid == 0 ? cm_hi : cm_lo) + fbase + (size_t)(kc + 1) * 512;
        stage_frag_async(g, lds0 + (unsigned)((buf ^ 1) * 2 + wid) * 1024u, lane);
        asm volatile("s_wait_asynccnt 0x2" ::: "memory");  // current fragment done
      } else {
        asm volatile("s_wait_asynccnt 0x0" ::: "memory");
      }
    }
    __syncthreads();  // publish sB[buf] to all 4 waves

    const int k0 = kc * 32 + (half ? 8 : 0);
    v8bf a0h = *(const v8bf*)(arh + k0);
    v8bf a1h = *(const v8bf*)(arh + k0 + 16);
    v8bf a0l = *(const v8bf*)(arl + k0);
    v8bf a1l = *(const v8bf*)(arl + k0 + 16);
    v16bf ah, al;
    #pragma unroll
    for (int i = 0; i < 8; ++i) { ah[i] = a0h[i]; ah[8 + i] = a1h[i]; al[i] = a0l[i]; al[8 + i] = a1l[i]; }

    v16bf bh = *(const v16bf*)&sB[buf][0][lane * 16];   // ds_load_b128 x2
    v16bf bl = *(const v16bf*)&sB[buf][1][lane * 16];
    acc = WMMA_BF16(ah, bh, acc);
    acc = WMMA_BF16(al, bh, acc);
    acc = WMMA_BF16(ah, bl, acc);

    __syncthreads();  // everyone done with sB[buf] before it is restaged
  }

  const int slot = (t + dm1 + 1) % NSLOT;
  float* dst = hdel + (size_t)slot * BB * HH;
  const int n = nt * 16 + (lane & 15);
  #pragma unroll
  for (int r = 0; r < 8; ++r) {
    const int m = wid * 16 + r + 8 * half;
    dst[(size_t)m * HH + n] += acc[r];
  }
}

// out = hdel[slot(T-1)] @ We^T + b_e
__global__ __launch_bounds__(32) void k_output(
    const float* __restrict__ hslot, const __bf16* __restrict__ we_hi, const __bf16* __restrict__ we_lo,
    const float* __restrict__ be, float* __restrict__ out) {
  const int nt = blockIdx.x;        // 0..15 (O tiles)
  const int mt = blockIdx.y;        // 0..3
  const int lane = threadIdx.x;
  const int half = lane >> 4;
  const int row = mt * 16 + (lane & 15);
  const float* arow = hslot + (size_t)row * HH;

  v8f acc = {};
  #pragma unroll
  for (int kc = 0; kc < 16; ++kc) {
    const int k0 = kc * 32 + (half ? 8 : 0);
    v4f f0 = *(const v4f*)(arow + k0);
    v4f f1 = *(const v4f*)(arow + k0 + 4);
    v4f f2 = *(const v4f*)(arow + k0 + 16);
    v4f f3 = *(const v4f*)(arow + k0 + 20);
    v16bf ah, al;
    #pragma unroll
    for (int i = 0; i < 4; ++i) { __bf16 h, l; split_bf16(f0[i], h, l); ah[i]      = h; al[i]      = l; }
    #pragma unroll
    for (int i = 0; i < 4; ++i) { __bf16 h, l; split_bf16(f1[i], h, l); ah[4 + i]  = h; al[4 + i]  = l; }
    #pragma unroll
    for (int i = 0; i < 4; ++i) { __bf16 h, l; split_bf16(f2[i], h, l); ah[8 + i]  = h; al[8 + i]  = l; }
    #pragma unroll
    for (int i = 0; i < 4; ++i) { __bf16 h, l; split_bf16(f3[i], h, l); ah[12 + i] = h; al[12 + i] = l; }
    const size_t boff = ((size_t)(nt * 16 + kc) * 32 + lane) * 16;
    v16bf bh = *(const v16bf*)(we_hi + boff);
    v16bf bl = *(const v16bf*)(we_lo + boff);
    acc = WMMA_BF16(ah, bh, acc);
    acc = WMMA_BF16(al, bh, acc);
    acc = WMMA_BF16(ah, bl, acc);
  }
  const int n = nt * 16 + (lane & 15);
  const float bias = be[n];
  #pragma unroll
  for (int r = 0; r < 8; ++r) {
    const int m = mt * 16 + r + 8 * half;
    out[(size_t)m * OO + n] = acc[r] + bias;
  }
}

// ---------------- host ----------------

extern "C" void kernel_launch(void* const* d_in, const int* in_sizes, int n_in,
                              void* d_out, int out_size, void* d_ws, size_t ws_size,
                              hipStream_t stream) {
  (void)in_sizes; (void)n_in; (void)out_size; (void)ws_size;
  const float* x   = (const float*)d_in[0];
  const float* wa  = (const float*)d_in[1];
  const float* ba  = (const float*)d_in[2];
  const float* lat = (const float*)d_in[3];
  const float* tau = (const float*)d_in[4];
  const float* we  = (const float*)d_in[5];
  const float* be  = (const float*)d_in[6];
  float* out = (float*)d_out;

  char* p = (char*)d_ws;                       // all offsets 256B-aligned
  __bf16* cm_hi = (__bf16*)(p);                //  8,388,608 B
  __bf16* cm_lo = (__bf16*)(p + 8388608);      //  8,388,608 B
  __bf16* wa_hi = (__bf16*)(p + 16777216);     //    262,144 B
  __bf16* wa_lo = (__bf16*)(p + 17039360);     //    262,144 B
  __bf16* we_hi = (__bf16*)(p + 17301504);     //    262,144 B
  __bf16* we_lo = (__bf16*)(p + 17563648);     //    262,144 B
  float*  hdel  = (float*) (p + 17825792);     //  2,228,224 B (17 x 64 x 512 f32)
  __bf16* h_hi  = (__bf16*)(p + 20054016);     //     65,536 B
  __bf16* h_lo  = (__bf16*)(p + 20119552);     //     65,536 B  (end: 20,185,088 B)

  k_zero<<<(NSLOT * BB * HH + 255) / 256, 256, 0, stream>>>(hdel, NSLOT * BB * HH);
  k_pack_wa<<<(32 * 8 * 32 * 16) / 256, 256, 0, stream>>>(wa, wa_hi, wa_lo);
  k_pack_we<<<(16 * 16 * 32 * 16) / 256, 256, 0, stream>>>(we, we_hi, we_lo);
  k_pack_cm<<<(16 * 32 * 16 * 32 * 16) / 256, 256, 0, stream>>>(lat, tau, cm_hi, cm_lo);

  // steps 0..T-2 (step T-1's h is never consumed; y[T-1] reads h_delayed before it)
  for (int t = 0; t < TT - 1; ++t) {
    k_afferent<<<dim3(32, 4), 32, 0, stream>>>(x, wa_hi, wa_lo, ba, hdel, h_hi, h_lo, t, t % NSLOT);
    int nd = (TT - 1) - t; if (nd > 16) nd = 16;   // only scatter to times <= T-1
    k_lateral<<<dim3(32, nd), 128, 0, stream>>>(h_hi, h_lo, cm_hi, cm_lo, hdel, t);
  }
  const int slotT = (TT - 1) % NSLOT;
  k_output<<<dim3(16, 4), 32, 0, stream>>>(hdel + (size_t)slotT * BB * HH, we_hi, we_lo, be, out);
}
